// MultiHeadAttention_63032940036651
// MI455X (gfx1250) — compile-verified, matrix operand order checked by reading, not run on hardware
//
#include <hip/hip_runtime.h>

// ---------------------------------------------------------------------------
// Multi-head attention for MI455X (gfx1250), wave32 + v_wmma_f32_16x16x32_f16.
// B=4, S=2048, D=1024, H=16, dh=64.  Outputs: out (B,S,D) then weights (B,H,S,S).
// ---------------------------------------------------------------------------

typedef _Float16 half_t;
typedef __attribute__((ext_vector_type(8)))  _Float16 v8h;
typedef __attribute__((ext_vector_type(16))) _Float16 v16h;
typedef __attribute__((ext_vector_type(8)))  float    v8f;
typedef int gv4i __attribute__((vector_size(16)));   // matches builtin's V4i

#define B_  4
#define S_  2048
#define D_  1024
#define H_  16
#define DH_ 64

constexpr float kScale = 0.125f;          // 1/sqrt(64)
constexpr float kNeg   = -9.0e15f;        // LARGE_NEG_F32

constexpr int BM  = 128;                  // block tile rows
constexpr int BN  = 64;                   // block tile cols
constexpr int BK  = 32;                   // K step (= WMMA K)
constexpr int AS  = BK + 8;               // LDS row stride, K-loop kernels (80B)
constexpr int AS2 = DH_ + 8;              // LDS row stride, scores kernel (144B)
constexpr int NT  = 256;                  // 8 wave32 waves per block

// --- gfx1250 async global->LDS copy (ASYNCcnt) with portable fallback -------
#if defined(__has_builtin)
#if __has_builtin(__builtin_amdgcn_global_load_async_to_lds_b128) && \
    __has_builtin(__builtin_amdgcn_s_wait_asynccnt)
#define USE_ASYNC_LDS 1
#endif
#endif

__device__ inline void async_copy16(const half_t* g, half_t* l) {
#ifdef USE_ASYNC_LDS
  __builtin_amdgcn_global_load_async_to_lds_b128(
      (__attribute__((address_space(1))) gv4i*)g,
      (__attribute__((address_space(3))) gv4i*)l, 0, 0);
#else
  *(float4*)l = *(const float4*)g;
#endif
}

__device__ inline void async_wait() {
#ifdef USE_ASYNC_LDS
  __builtin_amdgcn_s_wait_asynccnt(0);
#endif
}

// Load one 16x32 f16 WMMA operand fragment from LDS (row-major, stride STRIDE).
// ISA layout: lanes 0-15 -> K {0..7, 16..23}; lanes 16-31 -> K {8..15, 24..31}.
template <int STRIDE>
__device__ inline v16h frag_ld(const half_t* base, int rowBase, int koff, int lane) {
  int r  = rowBase + (lane & 15);
  int ko = koff + ((lane >> 4) << 3);
  const half_t* p = base + r * STRIDE + ko;
  v8h lo = *(const v8h*)(p);
  v8h hi = *(const v8h*)(p + 16);
  return __builtin_shufflevector(lo, hi, 0,1,2,3,4,5,6,7,8,9,10,11,12,13,14,15);
}

__device__ inline v8f wmma16(v16h a, v16h b, v8f c) {
  return __builtin_amdgcn_wmma_f32_16x16x32_f16(false, a, false, b, (short)0, c,
                                                false, false);
}

// ---------------------------------------------------------------------------
// Kernel 1: Q/K/V projection.  q = x @ W^T + b, stored f16.
// grid: (D/BN, B*S/BM, 3)   z selects Wq/Wk/Wv.  Register-pipelined staging.
// ---------------------------------------------------------------------------
__global__ __launch_bounds__(NT)
void qkv_kernel(const float* __restrict__ x,
                const float* __restrict__ Wq, const float* __restrict__ Wk,
                const float* __restrict__ Wv,
                const float* __restrict__ bq, const float* __restrict__ bk,
                const float* __restrict__ bv,
                half_t* __restrict__ Q, half_t* __restrict__ Kh,
                half_t* __restrict__ V) {
  const float* W; const float* bias; half_t* out;
  if      (blockIdx.z == 0) { W = Wq; bias = bq; out = Q;  }
  else if (blockIdx.z == 1) { W = Wk; bias = bk; out = Kh; }
  else                      { W = Wv; bias = bv; out = V;  }

  __shared__ __align__(16) half_t At[BM * AS];
  __shared__ __align__(16) half_t Bt[BN * AS];

  const int tid  = threadIdx.x;
  const int lane = tid & 31;
  const int wave = tid >> 5;
  const int wr   = wave & 3;       // 4 row-waves * 32 rows = 128
  const int wc   = wave >> 2;      // 2 col-waves * 32 cols = 64
  const int mBase = blockIdx.y * BM;
  const int nBase = blockIdx.x * BN;

  float4 fa[4], fb[2];
  auto loadA = [&](int k0) {
    #pragma unroll
    for (int u = 0; u < 4; u++) {
      int idx = tid + u * NT, row = idx >> 3, c4 = idx & 7;
      fa[u] = *(const float4*)(x + (size_t)(mBase + row) * D_ + k0 + c4 * 4);
    }
  };
  auto loadB = [&](int k0) {
    #pragma unroll
    for (int u = 0; u < 2; u++) {
      int idx = tid + u * NT, row = idx >> 3, c4 = idx & 7;
      fb[u] = *(const float4*)(W + (size_t)(nBase + row) * D_ + k0 + c4 * 4);
    }
  };
  auto storeLDS = [&]() {
    #pragma unroll
    for (int u = 0; u < 4; u++) {
      int idx = tid + u * NT, row = idx >> 3, c4 = idx & 7;
      half_t* d = &At[row * AS + c4 * 4];
      d[0] = (half_t)fa[u].x; d[1] = (half_t)fa[u].y;
      d[2] = (half_t)fa[u].z; d[3] = (half_t)fa[u].w;
    }
    #pragma unroll
    for (int u = 0; u < 2; u++) {
      int idx = tid + u * NT, row = idx >> 3, c4 = idx & 7;
      half_t* d = &Bt[row * AS + c4 * 4];
      d[0] = (half_t)fb[u].x; d[1] = (half_t)fb[u].y;
      d[2] = (half_t)fb[u].z; d[3] = (half_t)fb[u].w;
    }
  };

  v8f acc[2][2] = {};
  loadA(0); loadB(0);
  for (int k0 = 0; k0 < D_; k0 += BK) {
    storeLDS();
    __syncthreads();
    if (k0 + BK < D_) { loadA(k0 + BK); loadB(k0 + BK); }  // overlap with WMMA
    v16h a0 = frag_ld<AS>(At, wr * 32,      0, lane);
    v16h a1 = frag_ld<AS>(At, wr * 32 + 16, 0, lane);
    v16h b0 = frag_ld<AS>(Bt, wc * 32,      0, lane);
    v16h b1 = frag_ld<AS>(Bt, wc * 32 + 16, 0, lane);
    acc[0][0] = wmma16(a0, b0, acc[0][0]);
    acc[0][1] = wmma16(a0, b1, acc[0][1]);
    acc[1][0] = wmma16(a1, b0, acc[1][0]);
    acc[1][1] = wmma16(a1, b1, acc[1][1]);
    __syncthreads();
  }

  #pragma unroll
  for (int fm = 0; fm < 2; fm++)
    #pragma unroll
    for (int fn = 0; fn < 2; fn++) {
      int rowB = mBase + wr * 32 + fm * 16 + ((lane >> 4) << 3);
      int col  = nBase + wc * 32 + fn * 16 + (lane & 15);
      float bb = bias[col];
      #pragma unroll
      for (int r = 0; r < 8; r++)
        out[(size_t)(rowB + r) * D_ + col] = (half_t)(acc[fm][fn][r] + bb);
    }
}

// ---------------------------------------------------------------------------
// Kernel 2: logits = (Q_h @ K_h^T) * scale, causal + padding mask, f32.
// grid: (S/BN [j], S/BM [i], B*H).  Whole K=64 staged once (async->LDS),
// single barrier, 8 back-to-back WMMAs.  Fully-masked tiles skipped.
// ---------------------------------------------------------------------------
__global__ __launch_bounds__(NT)
void scores_kernel(const half_t* __restrict__ Q, const half_t* __restrict__ Kh,
                   const int* __restrict__ pad, float* __restrict__ wts) {
  const int bh = blockIdx.z, b = bh >> 4, h = bh & 15;
  const int iBase = blockIdx.y * BM;
  const int jBase = blockIdx.x * BN;
  if (jBase > iBase + (BM - 1)) return;   // entirely above the diagonal

  __shared__ __align__(16) half_t At[BM * AS2];   // 128 x 64 (+pad)
  __shared__ __align__(16) half_t Bt[BN * AS2];   // 64 x 64 (+pad)
  const int tid = threadIdx.x, lane = tid & 31, wave = tid >> 5;
  const int wr = wave & 3, wc = wave >> 2;

  const half_t* qrow = Q  + (size_t)b * S_ * D_ + h * DH_;
  const half_t* krow = Kh + (size_t)b * S_ * D_ + h * DH_;

  // Stage full Q tile (128x64) and K tile (64x64), 16B chunks, async to LDS.
  #pragma unroll
  for (int u = 0; u < 4; u++) {
    int idx = tid + u * NT;           // 0..1023 : 8 chunks per row
    int row = idx >> 3, c = idx & 7;
    async_copy16(qrow + (size_t)(iBase + row) * D_ + c * 8, &At[row * AS2 + c * 8]);
  }
  #pragma unroll
  for (int u = 0; u < 2; u++) {
    int idx = tid + u * NT;           // 0..511
    int row = idx >> 3, c = idx & 7;
    async_copy16(krow + (size_t)(jBase + row) * D_ + c * 8, &Bt[row * AS2 + c * 8]);
  }
  async_wait();
  __syncthreads();

  v8f acc[2][2] = {};
  #pragma unroll
  for (int ks = 0; ks < DH_; ks += BK) {
    v16h a0 = frag_ld<AS2>(At, wr * 32,      ks, lane);
    v16h a1 = frag_ld<AS2>(At, wr * 32 + 16, ks, lane);
    v16h b0 = frag_ld<AS2>(Bt, wc * 32,      ks, lane);
    v16h b1 = frag_ld<AS2>(Bt, wc * 32 + 16, ks, lane);
    acc[0][0] = wmma16(a0, b0, acc[0][0]);
    acc[0][1] = wmma16(a0, b1, acc[0][1]);
    acc[1][0] = wmma16(a1, b0, acc[1][0]);
    acc[1][1] = wmma16(a1, b1, acc[1][1]);
  }

  float* wrow = wts + (size_t)bh * S_ * S_;
  #pragma unroll
  for (int fm = 0; fm < 2; fm++)
    #pragma unroll
    for (int fn = 0; fn < 2; fn++) {
      int rowB = iBase + wr * 32 + fm * 16 + ((lane >> 4) << 3);
      int col  = jBase + wc * 32 + fn * 16 + (lane & 15);
      int pj   = pad[b * S_ + col];
      #pragma unroll
      for (int r = 0; r < 8; r++) {
        int i = rowB + r;
        float v = acc[fm][fn][r] * kScale;
        if (col > i || pj == 0) v = kNeg;
        wrow[(size_t)i * S_ + col] = v;
      }
    }
}

// ---------------------------------------------------------------------------
// Kernel 3: row softmax in place; zero-fill j > i (never reads upper triangle).
// grid: (B*H*S) blocks of 256; whole row lives in registers (8 per thread).
// ---------------------------------------------------------------------------
__global__ __launch_bounds__(NT)
void softmax_kernel(float* __restrict__ wts) {
  const int row = blockIdx.x;                 // bh*S + i
  const int i   = row & (S_ - 1);
  float* p = wts + (size_t)row * S_;
  const int tid = threadIdx.x;
  const int L   = i + 1;

  float v[S_ / NT];
  float m = -3.4e38f;
  #pragma unroll
  for (int it = 0; it < S_ / NT; it++) {
    int j = tid + it * NT;
    v[it] = (j < L) ? p[j] : -3.4e38f;
    m = fmaxf(m, v[it]);
  }
  __shared__ float red[NT];
  red[tid] = m; __syncthreads();
  for (int s = NT >> 1; s > 0; s >>= 1) {
    if (tid < s) red[tid] = fmaxf(red[tid], red[tid + s]);
    __syncthreads();
  }
  m = red[0]; __syncthreads();

  float sum = 0.f;
  #pragma unroll
  for (int it = 0; it < S_ / NT; it++) {
    int j = tid + it * NT;
    float e = (j < L) ? __expf(v[it] - m) : 0.f;
    v[it] = e; sum += e;
  }
  red[tid] = sum; __syncthreads();
  for (int s = NT >> 1; s > 0; s >>= 1) {
    if (tid < s) red[tid] += red[tid + s];
    __syncthreads();
  }
  float inv = 1.0f / red[0];
  #pragma unroll
  for (int it = 0; it < S_ / NT; it++) {
    int j = tid + it * NT;
    p[j] = (j < L) ? v[it] * inv : 0.f;
  }
}

// ---------------------------------------------------------------------------
// Kernel 4: ctx_h = weights @ V_h.  K-loop stops at the diagonal tile.
// grid: (1, S/BM [i], B*H).  Register-pipelined; V transposed into LDS.
// ---------------------------------------------------------------------------
__global__ __launch_bounds__(NT)
void av_kernel(const float* __restrict__ wts, const half_t* __restrict__ V,
               half_t* __restrict__ ctx) {
  const int bh = blockIdx.z, b = bh >> 4, h = bh & 15;
  const int iBase = blockIdx.y * BM;

  __shared__ __align__(16) half_t At[BM * AS];
  __shared__ __align__(16) half_t Bt[BN * AS];
  const int tid = threadIdx.x, lane = tid & 31, wave = tid >> 5;
  const int wr = wave & 3, wc = wave >> 2;

  const float*  wrow = wts + (size_t)bh * S_ * S_;
  const half_t* vrow = V + (size_t)b * S_ * D_ + h * DH_;
  const int jEnd = iBase + BM;               // weights are 0 for j > i

  float4 fa[4];
  v8h    vb;                                 // one 8-half chunk per thread
  const int vk = tid >> 3, vd8 = (tid & 7) * 8;
  auto loadA = [&](int k0) {
    #pragma unroll
    for (int u = 0; u < 4; u++) {
      int idx = tid + u * NT, row = idx >> 3, c4 = idx & 7;
      fa[u] = *(const float4*)(wrow + (size_t)(iBase + row) * S_ + k0 + c4 * 4);
    }
  };
  auto loadB = [&](int k0) {
    vb = *(const v8h*)(vrow + (size_t)(k0 + vk) * D_ + vd8);
  };
  auto storeLDS = [&]() {
    #pragma unroll
    for (int u = 0; u < 4; u++) {
      int idx = tid + u * NT, row = idx >> 3, c4 = idx & 7;
      half_t* d = &At[row * AS + c4 * 4];
      d[0] = (half_t)fa[u].x; d[1] = (half_t)fa[u].y;
      d[2] = (half_t)fa[u].z; d[3] = (half_t)fa[u].w;
    }
    #pragma unroll
    for (int e = 0; e < 8; e++) Bt[(vd8 + e) * AS + vk] = vb[e];  // transpose
  };

  v8f acc[2][2] = {};
  loadA(0); loadB(0);
  for (int k0 = 0; k0 < jEnd; k0 += BK) {
    storeLDS();
    __syncthreads();
    if (k0 + BK < jEnd) { loadA(k0 + BK); loadB(k0 + BK); }
    v16h a0 = frag_ld<AS>(At, wr * 32,      0, lane);
    v16h a1 = frag_ld<AS>(At, wr * 32 + 16, 0, lane);
    v16h b0 = frag_ld<AS>(Bt, wc * 32,      0, lane);
    v16h b1 = frag_ld<AS>(Bt, wc * 32 + 16, 0, lane);
    acc[0][0] = wmma16(a0, b0, acc[0][0]);
    acc[0][1] = wmma16(a0, b1, acc[0][1]);
    acc[1][0] = wmma16(a1, b0, acc[1][0]);
    acc[1][1] = wmma16(a1, b1, acc[1][1]);
    __syncthreads();
  }

  #pragma unroll
  for (int fm = 0; fm < 2; fm++)
    #pragma unroll
    for (int fn = 0; fn < 2; fn++) {
      int rowB = iBase + wr * 32 + fm * 16 + ((lane >> 4) << 3);
      int col  = wc * 32 + fn * 16 + (lane & 15);   // d in [0,64)
      #pragma unroll
      for (int r = 0; r < 8; r++)
        ctx[(size_t)(b * S_ + rowB + r) * D_ + h * DH_ + col] =
            (half_t)acc[fm][fn][r];
    }
}

// ---------------------------------------------------------------------------
// Kernel 5: out = ctx @ Wo^T + bo   (f16 A from ws, f32 result).
// grid: (D/BN, B*S/BM).  Register-pipelined staging.
// ---------------------------------------------------------------------------
__global__ __launch_bounds__(NT)
void oproj_kernel(const half_t* __restrict__ ctx, const float* __restrict__ Wo,
                  const float* __restrict__ bo, float* __restrict__ out) {
  __shared__ __align__(16) half_t At[BM * AS];
  __shared__ __align__(16) half_t Bt[BN * AS];
  const int tid = threadIdx.x, lane = tid & 31, wave = tid >> 5;
  const int wr = wave & 3, wc = wave >> 2;
  const int mBase = blockIdx.y * BM;
  const int nBase = blockIdx.x * BN;

  float4 ha[2];                 // raw f16 payload (8 halves per float4)
  float4 fb[2];
  auto loadA = [&](int k0) {
    #pragma unroll
    for (int u = 0; u < 2; u++) {
      int idx = tid + u * NT, row = idx >> 2, c = idx & 3;   // 4 chunks per row
      ha[u] = *(const float4*)(ctx + (size_t)(mBase + row) * D_ + k0 + c * 8);
    }
  };
  auto loadB = [&](int k0) {
    #pragma unroll
    for (int u = 0; u < 2; u++) {
      int idx = tid + u * NT, row = idx >> 3, c4 = idx & 7;
      fb[u] = *(const float4*)(Wo + (size_t)(nBase + row) * D_ + k0 + c4 * 4);
    }
  };
  auto storeLDS = [&]() {
    #pragma unroll
    for (int u = 0; u < 2; u++) {
      int idx = tid + u * NT, row = idx >> 2, c = idx & 3;
      *(float4*)&At[row * AS + c * 8] = ha[u];
    }
    #pragma unroll
    for (int u = 0; u < 2; u++) {
      int idx = tid + u * NT, row = idx >> 3, c4 = idx & 7;
      half_t* d = &Bt[row * AS + c4 * 4];
      d[0] = (half_t)fb[u].x; d[1] = (half_t)fb[u].y;
      d[2] = (half_t)fb[u].z; d[3] = (half_t)fb[u].w;
    }
  };

  v8f acc[2][2] = {};
  loadA(0); loadB(0);
  for (int k0 = 0; k0 < D_; k0 += BK) {
    storeLDS();
    __syncthreads();
    if (k0 + BK < D_) { loadA(k0 + BK); loadB(k0 + BK); }
    v16h a0 = frag_ld<AS>(At, wr * 32,      0, lane);
    v16h a1 = frag_ld<AS>(At, wr * 32 + 16, 0, lane);
    v16h b0 = frag_ld<AS>(Bt, wc * 32,      0, lane);
    v16h b1 = frag_ld<AS>(Bt, wc * 32 + 16, 0, lane);
    acc[0][0] = wmma16(a0, b0, acc[0][0]);
    acc[0][1] = wmma16(a0, b1, acc[0][1]);
    acc[1][0] = wmma16(a1, b0, acc[1][0]);
    acc[1][1] = wmma16(a1, b1, acc[1][1]);
    __syncthreads();
  }

  #pragma unroll
  for (int fm = 0; fm < 2; fm++)
    #pragma unroll
    for (int fn = 0; fn < 2; fn++) {
      int rowB = mBase + wr * 32 + fm * 16 + ((lane >> 4) << 3);
      int col  = nBase + wc * 32 + fn * 16 + (lane & 15);
      float bb = bo[col];
      #pragma unroll
      for (int r = 0; r < 8; r++)
        out[(size_t)(rowB + r) * D_ + col] = acc[fm][fn][r] + bb;
    }
}

// ---------------------------------------------------------------------------
extern "C" void kernel_launch(void* const* d_in, const int* in_sizes, int n_in,
                              void* d_out, int out_size, void* d_ws, size_t ws_size,
                              hipStream_t stream) {
  (void)in_sizes; (void)n_in; (void)out_size; (void)ws_size;
  const float* query = (const float*)d_in[0];
  const int*   pad   = (const int*)d_in[1];
  // d_in[2] = causal_mask (tril) -- applied analytically (col > row).
  const float* Wq = (const float*)d_in[3];  const float* bq = (const float*)d_in[4];
  const float* Wk = (const float*)d_in[5];  const float* bk = (const float*)d_in[6];
  const float* Wv = (const float*)d_in[7];  const float* bv = (const float*)d_in[8];
  const float* Wo = (const float*)d_in[9];  const float* bo = (const float*)d_in[10];

  float* out = (float*)d_out;
  float* wts = out + (size_t)B_ * S_ * D_;          // weights after out

  const size_t per = (size_t)B_ * S_ * D_;          // 8M halves each
  half_t* qh = (half_t*)d_ws;
  half_t* kh = qh + per;
  half_t* vh = kh + per;
  half_t* ch = vh + per;                            // context, f16

  qkv_kernel   <<<dim3(D_ / BN, (B_ * S_) / BM, 3), NT, 0, stream>>>(
      query, Wq, Wk, Wv, bq, bk, bv, qh, kh, vh);
  scores_kernel<<<dim3(S_ / BN, S_ / BM, B_ * H_), NT, 0, stream>>>(
      qh, kh, pad, wts);
  softmax_kernel<<<dim3(B_ * H_ * S_), NT, 0, stream>>>(wts);
  av_kernel    <<<dim3(1, S_ / BM, B_ * H_), NT, 0, stream>>>(wts, vh, ch);
  oproj_kernel <<<dim3(D_ / BN, (B_ * S_) / BM), NT, 0, stream>>>(ch, Wo, bo, out);
}
